// IsgnMeasNoteDecoder_42743514530126
// MI455X (gfx1250) — matrix operand order repaired and checked by reading, not compiled, not run on hardware
//
#include <hip/hip_runtime.h>

// ---------------- model dims ----------------
#define N_NOTES 2000
#define N_BEATS 400
#define N_MEAS  100
#define E_TYPES 10
#define NOTE_IN 256
#define ENC     64
#define TREG    32
#define OUTD    11
#define MARGIND 64
#define SEC     75           // OUTD + MARGIND
#define C_IN    461          // 256+2+64+64+11+64
#define FBH     322
#define SEQ_IT  2
#define GRAPH_IT 3
#define ML_IN   194

// ---------------- padded GEMM dims ----------------
#define KP_X   480    // C_IN -> mult of 32
#define NP_W   240    // 3*SEC=225 -> mult of 16 (and of 5*16)
#define KP_N   2016   // N_NOTES -> mult of 32
#define NP_UZR 160    // 150 -> mult of 16
#define KP_T   96     // SEC padded to mult of 32
#define NP_UH  80     // 75 -> mult of 16
#define NP_FC1 64

#define CDIV(a,b) (((a)+(b)-1)/(b))

typedef __attribute__((ext_vector_type(16))) __bf16 v16bf;
typedef __attribute__((ext_vector_type(8)))  __bf16 v8bf;
typedef __attribute__((ext_vector_type(8)))  float  v8f;

__device__ __forceinline__ float sigf(float x) { return 1.f / (1.f + __expf(-x)); }

__device__ __forceinline__ __bf16 f2bf(float f) {
  unsigned u = __float_as_uint(f);
  unsigned r = u + 0x7FFFu + ((u >> 16) & 1u);   // round-to-nearest-even
  unsigned short h = (unsigned short)(r >> 16);
  __bf16 out;
  __builtin_memcpy(&out, &h, 2);
  return out;
}

// ================= WMMA GEMM core =================
// A: row-major bf16, lda elements, K zero-padded to Kpad.
//    A-frag (contiguous k, as Bp): lanes 0-15 = rows, K {0..15}; lanes 16-31: K {16..31}
//    -> per lane exactly two contiguous 16B chunks of the row-major matrix.
// Bp: packed fragment-major bf16. Tile (kt,nt) occupies 512 elems at
//     (kt*ntn + nt)*512; element j of lane L == B[kt*32+(L>>4)*16+j, nt*16+(L&15)].
// Each wave computes a 16 x (NT*16) strip: one A-frag feeds NT WMMAs.

__device__ __forceinline__ v16bf load_afrag(const __bf16* __restrict__ p) {
  v8bf lo = *(const v8bf*)(p);
  v8bf hi = *(const v8bf*)(p + 8);
  v16bf a;
#pragma unroll
  for (int i = 0; i < 8; ++i) { a[i] = lo[i]; a[i + 8] = hi[i]; }
  return a;
}

template <int NT>
__global__ void k_wmma_gemmT(const __bf16* __restrict__ A, int lda, long long aStride,
                             const __bf16* __restrict__ Bp, long long bStride,
                             float* __restrict__ C, int ldc, long long cStride,
                             int M, int Kpad, int Npad) {
  int bat  = blockIdx.z;
  int wave = threadIdx.x >> 5;
  int lane = threadIdx.x & 31;
  int ntn  = Npad >> 4;
  int mtn  = M >> 4;
  int ngrp = ntn / NT;
  int grp  = blockIdx.x * (blockDim.x >> 5) + wave;
  if (grp >= mtn * ngrp) return;
  int mt = grp / ngrp, ng = grp % ngrp;
  int half = lane >> 4, r = lane & 15;
  const __bf16* Arow  = A + (size_t)bat * aStride + (size_t)(mt * 16 + r) * lda + half * 16;
  const __bf16* Bbase = Bp + (size_t)bat * bStride + ((size_t)ng * NT) * 512 + (size_t)lane * 16;
  v8f acc[NT];
  v8f zero = {0.f, 0.f, 0.f, 0.f, 0.f, 0.f, 0.f, 0.f};
#pragma unroll
  for (int i = 0; i < NT; ++i) acc[i] = zero;
  int nkt = Kpad >> 5;
  for (int kt = 0; kt < nkt; ++kt) {
    v16bf a = load_afrag(Arow + kt * 32);
    const __bf16* bk = Bbase + (size_t)kt * ntn * 512;
    if (kt + 1 < nkt) {
      __builtin_prefetch(Arow + (kt + 1) * 32, 0, 1);
      __builtin_prefetch(bk + (size_t)ntn * 512, 0, 1);
    }
#pragma unroll
    for (int i = 0; i < NT; ++i) {
      v16bf b = *(const v16bf*)(bk + (size_t)i * 512);
      acc[i] = __builtin_amdgcn_wmma_f32_16x16x32_bf16(false, a, false, b, (short)0,
                                                       acc[i], false, false);
    }
  }
  float* Crow = C + (size_t)bat * cStride + (size_t)(mt * 16 + half * 8) * ldc +
                (size_t)(ng * NT) * 16 + r;
#pragma unroll
  for (int i = 0; i < NT; ++i)
#pragma unroll
    for (int v = 0; v < 8; ++v) Crow[i * 16 + (size_t)v * ldc] = acc[i][v];
}

// reduction variant: accumulates over nbatch (A,B) pairs into one C
template <int NT>
__global__ void k_wmma_gemm_redT(const __bf16* __restrict__ A, int lda, long long aStride,
                                 const __bf16* __restrict__ Bp, long long bStride,
                                 float* __restrict__ C, int ldc,
                                 int M, int Kpad, int Npad, int nbatch) {
  int wave = threadIdx.x >> 5;
  int lane = threadIdx.x & 31;
  int ntn  = Npad >> 4;
  int mtn  = M >> 4;
  int ngrp = ntn / NT;
  int grp  = blockIdx.x * (blockDim.x >> 5) + wave;
  if (grp >= mtn * ngrp) return;
  int mt = grp / ngrp, ng = grp % ngrp;
  int half = lane >> 4, r = lane & 15;
  v8f acc[NT];
  v8f zero = {0.f, 0.f, 0.f, 0.f, 0.f, 0.f, 0.f, 0.f};
#pragma unroll
  for (int i = 0; i < NT; ++i) acc[i] = zero;
  int nkt = Kpad >> 5;
  for (int bat = 0; bat < nbatch; ++bat) {
    const __bf16* Arow  = A + (size_t)bat * aStride + (size_t)(mt * 16 + r) * lda + half * 16;
    const __bf16* Bbase = Bp + (size_t)bat * bStride + ((size_t)ng * NT) * 512 +
                          (size_t)lane * 16;
    for (int kt = 0; kt < nkt; ++kt) {
      v16bf a = load_afrag(Arow + kt * 32);
      const __bf16* bk = Bbase + (size_t)kt * ntn * 512;
      if (kt + 1 < nkt) {
        __builtin_prefetch(Arow + (kt + 1) * 32, 0, 1);
        __builtin_prefetch(bk + (size_t)ntn * 512, 0, 1);
      }
#pragma unroll
      for (int i = 0; i < NT; ++i) {
        v16bf b = *(const v16bf*)(bk + (size_t)i * 512);
        acc[i] = __builtin_amdgcn_wmma_f32_16x16x32_bf16(false, a, false, b, (short)0,
                                                         acc[i], false, false);
      }
    }
  }
  float* Crow = C + (size_t)(mt * 16 + half * 8) * ldc + (size_t)(ng * NT) * 16 + r;
#pragma unroll
  for (int i = 0; i < NT; ++i)
#pragma unroll
    for (int v = 0; v < 8; ++v) Crow[i * 16 + (size_t)v * ldc] = acc[i][v];
}

// GEMM variant that stores D directly as packed bf16 B-fragments for the next GEMM
// (C layout lane->(m=half*8+v, n) maps to B layout as one contiguous v8bf per lane:
//  k-tile = mt>>1, lane L2 = (mt&1)*16 + n, elements j = half*8 .. half*8+7)
template <int NT>
__global__ void k_wmma_gemm_packout(const __bf16* __restrict__ A, int lda,
                                    const __bf16* __restrict__ Bp, long long bStride,
                                    __bf16* __restrict__ Dp, long long dStride,
                                    int M, int Kpad, int Npad) {
  int bat  = blockIdx.z;
  int wave = threadIdx.x >> 5;
  int lane = threadIdx.x & 31;
  int ntn  = Npad >> 4;
  int mtn  = M >> 4;
  int ngrp = ntn / NT;
  int grp  = blockIdx.x * (blockDim.x >> 5) + wave;
  if (grp >= mtn * ngrp) return;
  int mt = grp / ngrp, ng = grp % ngrp;
  int half = lane >> 4, r = lane & 15;
  const __bf16* Arow  = A + (size_t)(mt * 16 + r) * lda + half * 16;
  const __bf16* Bbase = Bp + (size_t)bat * bStride + ((size_t)ng * NT) * 512 +
                        (size_t)lane * 16;
  v8f acc[NT];
  v8f zero = {0.f, 0.f, 0.f, 0.f, 0.f, 0.f, 0.f, 0.f};
#pragma unroll
  for (int i = 0; i < NT; ++i) acc[i] = zero;
  int nkt = Kpad >> 5;
  for (int kt = 0; kt < nkt; ++kt) {
    v16bf a = load_afrag(Arow + kt * 32);
    const __bf16* bk = Bbase + (size_t)kt * ntn * 512;
#pragma unroll
    for (int i = 0; i < NT; ++i) {
      v16bf b = *(const v16bf*)(bk + (size_t)i * 512);
      acc[i] = __builtin_amdgcn_wmma_f32_16x16x32_bf16(false, a, false, b, (short)0,
                                                       acc[i], false, false);
    }
  }
  int L2 = ((mt & 1) << 4) + r;
#pragma unroll
  for (int i = 0; i < NT; ++i) {
    int nt = ng * NT + i;
    v8bf o;
#pragma unroll
    for (int v = 0; v < 8; ++v) o[v] = f2bf(acc[i][v]);
    __bf16* dp = Dp + (size_t)bat * dStride +
                 (((size_t)(mt >> 1) * ntn + nt) * 512) + (size_t)L2 * 16 + half * 8;
    *(v8bf*)dp = o;
  }
}

// pack a row-major f32 (K x N) matrix (batched) into WMMA-B bf16 fragments
__global__ void k_packB(const float* __restrict__ src, __bf16* __restrict__ dst,
                        int K, int N, int Kpad, int Npad, int batch,
                        long long srcStride, long long dstStride) {
  long long total = (long long)Kpad * Npad * batch;
  for (long long idx = blockIdx.x * (long long)blockDim.x + threadIdx.x; idx < total;
       idx += (long long)gridDim.x * blockDim.x) {
    long long per = (long long)Kpad * Npad;
    int  b  = (int)(idx / per);
    long long t = idx % per;
    long long tile = t >> 9;
    int inT = (int)(t & 511);
    int lane = inT >> 4, j = inT & 15;
    int ntn = Npad >> 4;
    int kt = (int)(tile / ntn), nt = (int)(tile % ntn);
    int k = kt * 32 + (lane >> 4) * 16 + j;
    int n = nt * 16 + (lane & 15);
    float v = (k < K && n < N) ? src[(long long)b * srcStride + (long long)k * N + n] : 0.f;
    dst[(long long)b * dstStride + t] = f2bf(v);
  }
}

__global__ void k_zero_bf16(__bf16* __restrict__ dst, long long count) {
  for (long long i = blockIdx.x * (long long)blockDim.x + threadIdx.x; i < count;
       i += (long long)gridDim.x * blockDim.x)
    dst[i] = f2bf(0.f);
}

// ============ edges transpose+convert: edT[e][m][k] = bf16(edges[e][k][m]) ============
__global__ void k_edgesT(const float* __restrict__ edges, __bf16* __restrict__ edT) {
  __shared__ float tile[16][17];
  int e = blockIdx.z;
  int k0 = blockIdx.y * 16, m0 = blockIdx.x * 16;
  tile[threadIdx.y][threadIdx.x] =
      edges[((long long)e * N_NOTES + (k0 + threadIdx.y)) * N_NOTES + m0 + threadIdx.x];
  __syncthreads();
  edT[((long long)e * N_NOTES + (m0 + threadIdx.y)) * KP_N + k0 + threadIdx.x] =
      f2bf(tile[threadIdx.x][threadIdx.y]);
}

__global__ void k_edT_pad(__bf16* __restrict__ edT) {
  long long total = (long long)E_TYPES * N_NOTES * (KP_N - N_NOTES);
  for (long long i = blockIdx.x * (long long)blockDim.x + threadIdx.x; i < total;
       i += (long long)gridDim.x * blockDim.x) {
    int pad = KP_N - N_NOTES;
    long long em = i / pad;
    int j = (int)(i % pad);
    edT[em * KP_N + N_NOTES + j] = f2bf(0.f);
  }
}

// ============ small setup kernels ============
__global__ void k_perform_z(const float* __restrict__ pe, const float* __restrict__ W,
                            const float* __restrict__ b, float* __restrict__ pz) {
  int t = threadIdx.x;  // 64
  float s = b[t];
  for (int k = 0; k < 16; ++k) s += pe[k] * W[k * ENC + t];
  pz[t] = fmaxf(s, 0.f);
}

__global__ void k_bias_s(const float* __restrict__ ba, const float* __restrict__ w,
                         float* __restrict__ bias) {
  int s = threadIdx.x;  // 256 threads
  if (s >= NP_W) return;
  float acc = 0.f;
  if (s < 3 * SEC) {
    for (int e = 0; e < E_TYPES; ++e)
      for (int c = 0; c < C_IN; ++c)
        acc += ba[c] * w[((long long)e * C_IN + c) * (3 * SEC) + s];
  }
  bias[s] = acc;
}

// sequential measure LSTM (100 steps, two cells), one block of 256 threads
__global__ void k_measure_lstm(const float* __restrict__ pz, const float* __restrict__ meas,
                               const float* __restrict__ wihF, const float* __restrict__ whhF,
                               const float* __restrict__ bF,
                               const float* __restrict__ wihB, const float* __restrict__ whhB,
                               const float* __restrict__ bB,
                               const float* __restrict__ mfcW, const float* __restrict__ mfcB,
                               float* __restrict__ mtv, float* __restrict__ mtv_out) {
  __shared__ float inp[ML_IN], gates[512], h[128], c[128];
  int t = threadIdx.x;
  if (t < 128) { h[t] = 0.f; c[t] = 0.f; }
  if (t < 2) inp[192 + t] = 0.f;
  __syncthreads();
  for (int s = 0; s < N_MEAS; ++s) {
    if (t < 64) inp[t] = pz[t];
    else if (t < 192) inp[t] = meas[s * 128 + (t - 64)];
    __syncthreads();
    {
      float gf = bF[t], gb = bB[t];
      for (int k = 0; k < ML_IN; ++k) {
        float iv = inp[k];
        gf += wihF[t * ML_IN + k] * iv;
        gb += wihB[t * ML_IN + k] * iv;
      }
      for (int k = 0; k < 64; ++k) {
        gf += whhF[t * 64 + k] * h[k];
        gb += whhB[t * 64 + k] * h[64 + k];
      }
      gates[t] = gf;
      gates[256 + t] = gb;
    }
    __syncthreads();
    if (t < 128) {
      int cell = t >> 6, u = t & 63, base = cell * 256;
      float ig = sigf(gates[base + u]), fg = sigf(gates[base + 64 + u]);
      float gg = tanhf(gates[base + 128 + u]), og = sigf(gates[base + 192 + u]);
      float cn = fg * c[t] + ig * gg;
      c[t] = cn;
      h[t] = og * tanhf(cn);
    }
    __syncthreads();
    if (t < 2) {
      float s2 = mfcB[t];
      for (int k = 0; k < 128; ++k) s2 += h[k] * mfcW[k * 2 + t];
      mtv[s * 2 + t] = s2;
      mtv_out[s * 2 + t] = s2;
      inp[192 + t] = s2;
    }
    __syncthreads();
  }
}

// builds x rows: [note_cat(258) | pz(64) | zeros(64) | init_out(11) | zeros(64)]
__global__ void k_build_x(const float* __restrict__ note, const float* __restrict__ mtv,
                          const int* __restrict__ measNum, const float* __restrict__ pz,
                          const float* __restrict__ w1, const float* __restrict__ b1,
                          const float* __restrict__ w2, const float* __restrict__ b2,
                          float* __restrict__ x) {
  int n = blockIdx.x, t = threadIdx.x;  // 64 threads
  __shared__ float nc[258], hid[64], init[16];
  for (int k = t; k < 256; k += 64) nc[k] = note[(long long)n * 256 + k];
  if (t < 2) nc[256 + t] = mtv[measNum[n] * 2 + t];
  __syncthreads();
  {
    float s = b1[t];
    for (int k = 0; k < 258; ++k) s += nc[k] * w1[k * 64 + t];
    hid[t] = fmaxf(s, 0.f);
  }
  __syncthreads();
  if (t < OUTD) {
    float s = b2[t];
    for (int k = 0; k < 64; ++k) s += hid[k] * w2[k * OUTD + t];
    init[t] = s;
  }
  __syncthreads();
  float* xr = x + (long long)n * C_IN;
  for (int cidx = t; cidx < C_IN; cidx += 64) {
    float v;
    if (cidx < 258)       v = nc[cidx];
    else if (cidx < 322)  v = pz[cidx - 258];
    else if (cidx < 386)  v = 0.f;
    else if (cidx < 397)  v = init[cidx - 386];
    else                  v = 0.f;
    xr[cidx] = v;
  }
}

__global__ void k_cvt_x(const float* __restrict__ x, __bf16* __restrict__ xb) {
  long long total = (long long)N_NOTES * KP_X;
  for (long long i = blockIdx.x * (long long)blockDim.x + threadIdx.x; i < total;
       i += (long long)gridDim.x * blockDim.x) {
    int n = (int)(i / KP_X), k = (int)(i % KP_X);
    xb[i] = (k < C_IN) ? f2bf(x[(long long)n * C_IN + k]) : f2bf(0.f);
  }
}

// ============ GRU elementwise kernels ============
__global__ void k_gru1(const float* __restrict__ wzrh, const float* __restrict__ bias,
                       const float* __restrict__ uzro, const float* __restrict__ x,
                       float* __restrict__ zb, __bf16* __restrict__ rx) {
  long long total = (long long)N_NOTES * KP_T;
  for (long long i = blockIdx.x * (long long)blockDim.x + threadIdx.x; i < total;
       i += (long long)gridDim.x * blockDim.x) {
    int n = (int)(i / KP_T), s = (int)(i % KP_T);
    if (s < SEC) {
      float z = sigf(wzrh[n * NP_W + s] + bias[s] + uzro[n * NP_UZR + s]);
      float r = sigf(wzrh[n * NP_W + SEC + s] + bias[SEC + s] + uzro[n * NP_UZR + SEC + s]);
      float xt = x[(long long)n * C_IN + (C_IN - SEC) + s];
      zb[i] = z;
      rx[i] = f2bf(r * xt);
    } else {
      zb[i] = 0.f;
      rx[i] = f2bf(0.f);
    }
  }
}

__global__ void k_gru2(const float* __restrict__ wzrh, const float* __restrict__ bias,
                       const float* __restrict__ rh, const float* __restrict__ zb,
                       float* __restrict__ x) {
  long long total = (long long)N_NOTES * SEC;
  for (long long i = blockIdx.x * (long long)blockDim.x + threadIdx.x; i < total;
       i += (long long)gridDim.x * blockDim.x) {
    int n = (int)(i / SEC), s = (int)(i % SEC);
    float h = tanhf(wzrh[n * NP_W + 2 * SEC + s] + bias[2 * SEC + s] + rh[n * KP_T + s]);
    float z = zb[n * KP_T + s];
    long long xi = (long long)n * C_IN + (C_IN - SEC) + s;
    x[xi] = (1.f - z) * x[xi] + z * h;
  }
}

// ============ beat context attention ============
__global__ void k_att_out(const float* __restrict__ x, const float* __restrict__ W,
                          const float* __restrict__ b, const float* __restrict__ cv,
                          const float* __restrict__ res, float* __restrict__ bc) {
  int bt = blockIdx.x, t = threadIdx.x;  // 64 threads
  __shared__ float o[5][11], a[5][11], sim[5], wn[5];
  for (int i = t; i < 55; i += 64) {
    int note = bt * 5 + i / 11, c2 = i % 11;
    o[i / 11][c2] = x[(long long)note * C_IN + 386 + c2];
  }
  __syncthreads();
  if (t < 55) {
    int i = t / 11, c2 = t % 11;
    float s = b[c2];
    for (int k = 0; k < 11; ++k) s += o[i][k] * W[k * 11 + c2];
    a[i][c2] = s;
  }
  __syncthreads();
  if (t < 5) {
    float s = 0.f;
    for (int k = 0; k < 11; ++k) s += a[t][k] * cv[k];
    sim[t] = s;
  }
  __syncthreads();
  if (t == 0) {
    float m = sim[0];
    for (int i = 1; i < 5; ++i) m = fmaxf(m, sim[i]);
    float d = 0.f;
    for (int i = 0; i < 5; ++i) { wn[i] = __expf(sim[i] - m); d += wn[i]; }
    for (int i = 0; i < 5; ++i) wn[i] /= d;
  }
  __syncthreads();
  if (t < 11) {
    float s = 0.f;
    for (int i = 0; i < 5; ++i) s += o[i][t] * wn[i];
    bc[bt * 96 + t] = s;
  }
  if (t >= 11 && t < 19) bc[bt * 96 + 75 + (t - 11)] = res[bt * 8 + (t - 11)];
}

__global__ void k_att_margin(const float* __restrict__ x, const float* __restrict__ W,
                             const float* __restrict__ b, const float* __restrict__ cv,
                             float* __restrict__ bc) {
  int bt = blockIdx.x, t = threadIdx.x;  // 64 threads
  __shared__ float m[5][64], a[5][64], sim[5][8], wn[5][8];
  for (int i = 0; i < 5; ++i) m[i][t] = x[(long long)(bt * 5 + i) * C_IN + 397 + t];
  __syncthreads();
  for (int i = 0; i < 5; ++i) {
    float s = b[t];
    for (int k = 0; k < 64; ++k) s += m[i][k] * W[k * 64 + t];
    a[i][t] = s;
  }
  __syncthreads();
  if (t < 40) {
    int i = t / 8, h = t % 8;
    float s = 0.f;
    for (int d = 0; d < 8; ++d) s += a[i][h * 8 + d] * cv[h * 8 + d];
    sim[i][h] = s;
  }
  __syncthreads();
  if (t < 8) {
    float mx = sim[0][t];
    for (int i = 1; i < 5; ++i) mx = fmaxf(mx, sim[i][t]);
    float d = 0.f;
    for (int i = 0; i < 5; ++i) { wn[i][t] = __expf(sim[i][t] - mx); d += wn[i][t]; }
    for (int i = 0; i < 5; ++i) wn[i][t] /= d;
  }
  __syncthreads();
  {
    int h = t >> 3;
    float s = 0.f;
    for (int i = 0; i < 5; ++i) s += m[i][t] * wn[i][h];
    bc[bt * 96 + 11 + t] = s;
  }
}

// bidirectional 32-hidden LSTM over T steps; one block, 256 threads
__global__ void k_bilstm(const float* __restrict__ in, int ldin, int din, int T,
                         const float* __restrict__ wihF, const float* __restrict__ whhF,
                         const float* __restrict__ bF,
                         const float* __restrict__ wihB, const float* __restrict__ whhB,
                         const float* __restrict__ bB, float* __restrict__ out) {
  __shared__ float inF[96], inB[96], h[64], c[64], g[256];
  int t = threadIdx.x;
  if (t < 64) { h[t] = 0.f; c[t] = 0.f; }
  __syncthreads();
  for (int s = 0; s < T; ++s) {
    for (int k = t; k < din; k += 256) {
      inF[k] = in[(long long)s * ldin + k];
      inB[k] = in[(long long)(T - 1 - s) * ldin + k];
    }
    __syncthreads();
    {
      int dir = t >> 7, j = t & 127;
      const float* wih = dir ? wihB : wihF;
      const float* whh = dir ? whhB : whhF;
      const float* bb  = dir ? bB : bF;
      const float* iv  = dir ? inB : inF;
      float s2 = bb[j];
      for (int k = 0; k < din; ++k) s2 += wih[j * din + k] * iv[k];
      for (int k = 0; k < 32; ++k) s2 += whh[j * 32 + k] * h[dir * 32 + k];
      g[t] = s2;
    }
    __syncthreads();
    if (t < 64) {
      int dir = t >> 5, u = t & 31, base = dir * 128;
      float ig = sigf(g[base + u]), fg = sigf(g[base + 32 + u]);
      float gg = tanhf(g[base + 64 + u]), og = sigf(g[base + 96 + u]);
      float cn = fg * c[t] + ig * gg;
      c[t] = cn;
      float hn = og * tanhf(cn);
      h[t] = hn;
      int row = dir ? (T - 1 - s) : s;
      out[(long long)row * 64 + dir * 32 + u] = hn;
    }
    __syncthreads();
  }
}

__global__ void k_tempo(const float* __restrict__ h2, const float* __restrict__ w,
                        const float* __restrict__ b, float* __restrict__ tempo) {
  int i = blockIdx.x * blockDim.x + threadIdx.x;
  if (i >= N_BEATS) return;
  float s = b[0];
  for (int k = 0; k < 64; ++k) s += h2[i * 64 + k] * w[k];
  tempo[i] = s;
}

__global__ void k_update_mid(const int* __restrict__ beat, const float* __restrict__ h2,
                             float* __restrict__ x) {
  long long total = (long long)N_NOTES * 64;
  for (long long i = blockIdx.x * (long long)blockDim.x + threadIdx.x; i < total;
       i += (long long)gridDim.x * blockDim.x) {
    int n = (int)(i / 64), j = (int)(i % 64);
    x[(long long)n * C_IN + FBH + j] = h2[beat[n] * 64 + j];
  }
}

__global__ void k_other(const float* __restrict__ hid, const float* __restrict__ b1,
                        const float* __restrict__ w2, const float* __restrict__ b2,
                        const float* __restrict__ tempo, const int* __restrict__ beat,
                        float* __restrict__ x, float* __restrict__ dout) {
  int n = blockIdx.x, t = threadIdx.x;  // 64 threads
  __shared__ float hdn[64];
  hdn[t] = fmaxf(hid[(long long)n * 64 + t] + b1[t], 0.f);
  __syncthreads();
  if (t < 10) {
    float s = b2[t];
    for (int k = 0; k < 64; ++k) s += hdn[k] * w2[k * 10 + t];
    x[(long long)n * C_IN + 387 + t] = s;
    dout[(long long)n * OUTD + 1 + t] = s;
  }
  if (t == 32) {
    float f0 = tempo[beat[n]];
    x[(long long)n * C_IN + 386] = f0;
    dout[(long long)n * OUTD] = f0;
  }
}

// ================= host launcher =================
extern "C" void kernel_launch(void* const* d_in, const int* in_sizes, int n_in,
                              void* d_out, int out_size, void* d_ws, size_t ws_size,
                              hipStream_t stream) {
  (void)in_sizes; (void)n_in; (void)out_size; (void)ws_size;
  const float* note_out    = (const float*)d_in[0];
  const float* measure_out = (const float*)d_in[1];
  const float* perf_emb    = (const float*)d_in[2];
  const float* res_info    = (const float*)d_in[3];
  const float* edges       = (const float*)d_in[4];
  const int*   beat_num    = (const int*)d_in[5];
  const int*   meas_num    = (const int*)d_in[6];
  const float* sve_w  = (const float*)d_in[7];
  const float* sve_b  = (const float*)d_in[8];
  const float* irf_w1 = (const float*)d_in[9];
  const float* irf_b1 = (const float*)d_in[10];
  const float* irf_w2 = (const float*)d_in[11];
  const float* irf_b2 = (const float*)d_in[12];
  const float* ml_f_wih = (const float*)d_in[13];
  const float* ml_f_whh = (const float*)d_in[14];
  const float* ml_f_b   = (const float*)d_in[15];
  const float* ml_b_wih = (const float*)d_in[16];
  const float* ml_b_whh = (const float*)d_in[17];
  const float* ml_b_b   = (const float*)d_in[18];
  const float* mfc_w = (const float*)d_in[19];
  const float* mfc_b = (const float*)d_in[20];
  const float* gg_ba  = (const float*)d_in[21];
  const float* gg_w   = (const float*)d_in[22];
  const float* gg_uzr = (const float*)d_in[23];
  const float* gg_uh  = (const float*)d_in[24];
  const float* tr0f_wih = (const float*)d_in[25];
  const float* tr0f_whh = (const float*)d_in[26];
  const float* tr0f_b   = (const float*)d_in[27];
  const float* tr0b_wih = (const float*)d_in[28];
  const float* tr0b_whh = (const float*)d_in[29];
  const float* tr0b_b   = (const float*)d_in[30];
  const float* tr1f_wih = (const float*)d_in[31];
  const float* tr1f_whh = (const float*)d_in[32];
  const float* tr1f_b   = (const float*)d_in[33];
  const float* tr1b_wih = (const float*)d_in[34];
  const float* tr1b_whh = (const float*)d_in[35];
  const float* tr1b_b   = (const float*)d_in[36];
  const float* fba_w  = (const float*)d_in[37];
  const float* fba_b  = (const float*)d_in[38];
  const float* fba_cv = (const float*)d_in[39];
  const float* fma_w  = (const float*)d_in[40];
  const float* fma_b  = (const float*)d_in[41];
  const float* fma_cv = (const float*)d_in[42];
  const float* tfc_w  = (const float*)d_in[43];
  const float* tfc_b  = (const float*)d_in[44];
  const float* fc_w1  = (const float*)d_in[45];
  const float* fc_b1  = (const float*)d_in[46];
  const float* fc_w2  = (const float*)d_in[47];
  const float* fc_b2  = (const float*)d_in[48];

  float* out = (float*)d_out;  // [2000*11 final_out][100*2 mtv]

  // ---- workspace bump allocator ----
  char* p = (char*)d_ws;
  auto alloc = [&](size_t bytes) -> void* {
    void* r = (void*)p;
    p += (bytes + 255) & ~(size_t)255;
    return r;
  };
  float*  X     = (float*) alloc((size_t)N_NOTES * C_IN * 4);
  __bf16* XB    = (__bf16*)alloc((size_t)N_NOTES * KP_X * 2);
  __bf16* EDT   = (__bf16*)alloc((size_t)E_TYPES * N_NOTES * KP_N * 2);
  __bf16* WP    = (__bf16*)alloc((size_t)E_TYPES * KP_X * NP_W * 2);
  __bf16* UZRP  = (__bf16*)alloc((size_t)KP_X * NP_UZR * 2);
  __bf16* UHP   = (__bf16*)alloc((size_t)KP_T * NP_UH * 2);
  __bf16* FC1P  = (__bf16*)alloc((size_t)KP_X * NP_FC1 * 2);
  float*  BIAS  = (float*) alloc(NP_W * 4);
  float*  PZ    = (float*) alloc(ENC * 4);
  float*  MTV   = (float*) alloc(N_MEAS * 2 * 4);
  __bf16* YB    = (__bf16*)alloc((size_t)E_TYPES * KP_N * NP_W * 2);  // packed B frags
  float*  WZRH  = (float*) alloc((size_t)N_NOTES * NP_W * 4);
  float*  UZRO  = (float*) alloc((size_t)N_NOTES * NP_UZR * 4);
  float*  ZB    = (float*) alloc((size_t)N_NOTES * KP_T * 4);
  __bf16* RX    = (__bf16*)alloc((size_t)N_NOTES * KP_T * 2);
  float*  RH    = (float*) alloc((size_t)N_NOTES * KP_T * 4);
  float*  BC    = (float*) alloc((size_t)N_BEATS * 96 * 4);
  float*  H1    = (float*) alloc((size_t)N_BEATS * 64 * 4);
  float*  H2    = (float*) alloc((size_t)N_BEATS * 64 * 4);
  float*  TMPO  = (float*) alloc((size_t)N_BEATS * 4);
  float*  HID   = (float*) alloc((size_t)N_NOTES * 64 * 4);

  // ---- setup ----
  k_perform_z<<<1, 64, 0, stream>>>(perf_emb, sve_w, sve_b, PZ);
  k_measure_lstm<<<1, 256, 0, stream>>>(PZ, measure_out, ml_f_wih, ml_f_whh, ml_f_b,
                                        ml_b_wih, ml_b_whh, ml_b_b, mfc_w, mfc_b, MTV,
                                        out + (size_t)N_NOTES * OUTD);
  k_build_x<<<N_NOTES, 64, 0, stream>>>(note_out, MTV, meas_num, PZ, irf_w1, irf_b1,
                                        irf_w2, irf_b2, X);
  {
    dim3 g(N_NOTES / 16, N_NOTES / 16, E_TYPES), b(16, 16, 1);
    k_edgesT<<<g, b, 0, stream>>>(edges, EDT);
    k_edT_pad<<<1250, 256, 0, stream>>>(EDT);
  }
  k_packB<<<2048, 256, 0, stream>>>(gg_w, WP, C_IN, 3 * SEC, KP_X, NP_W, E_TYPES,
                                    (long long)C_IN * 3 * SEC, (long long)KP_X * NP_W);
  k_packB<<<512, 256, 0, stream>>>(gg_uzr, UZRP, C_IN, 150, KP_X, NP_UZR, 1, 0, 0);
  k_packB<<<64, 256, 0, stream>>>(gg_uh, UHP, SEC, SEC, KP_T, NP_UH, 1, 0, 0);
  k_packB<<<256, 256, 0, stream>>>(fc_w1, FC1P, C_IN, 64, KP_X, NP_FC1, 1, 0, 0);
  k_bias_s<<<1, 256, 0, stream>>>(gg_ba, gg_w, BIAS);
  // zero YB once per launch so the K-pad rows (2000..2015) stay zero under packout
  k_zero_bf16<<<4096, 256, 0, stream>>>(YB, (long long)E_TYPES * KP_N * NP_W);

  const int mtiles = N_NOTES / 16;  // 125
  for (int si = 0; si < SEQ_IT; ++si) {
    // ---- gated graph: 3 iterations ----
    for (int gi = 0; gi < GRAPH_IT; ++gi) {
      k_cvt_x<<<3750, 256, 0, stream>>>(X, XB);
      // y_e = xb @ w_e  (batched over e), output directly as packed B fragments
      {
        int grps = mtiles * ((NP_W / 16) / 5);  // 125*3
        dim3 g(CDIV(grps, 4), 1, E_TYPES);
        k_wmma_gemm_packout<5><<<g, 128, 0, stream>>>(
            XB, KP_X, WP, (long long)KP_X * NP_W, YB, (long long)KP_N * NP_W, N_NOTES,
            KP_X, NP_W);
      }
      // wzrh = sum_e edgesT_e @ y_e
      {
        int grps = mtiles * ((NP_W / 16) / 5);
        k_wmma_gemm_redT<5><<<CDIV(grps, 4), 128, 0, stream>>>(
            EDT, KP_N, (long long)N_NOTES * KP_N, YB, (long long)KP_N * NP_W, WZRH, NP_W,
            N_NOTES, KP_N, NP_W, E_TYPES);
      }
      // uzr = x @ gg_uzr
      {
        int grps = mtiles * ((NP_UZR / 16) / 5);  // 125*2
        k_wmma_gemmT<5><<<dim3(CDIV(grps, 4), 1, 1), 128, 0, stream>>>(
            XB, KP_X, 0, UZRP, 0, UZRO, NP_UZR, 0, N_NOTES, KP_X, NP_UZR);
      }
      k_gru1<<<750, 256, 0, stream>>>(WZRH, BIAS, UZRO, X, ZB, RX);
      // rh = (r*x_tail) @ uh
      {
        int grps = mtiles * ((NP_UH / 16) / 5);  // 125
        k_wmma_gemmT<5><<<dim3(CDIV(grps, 4), 1, 1), 128, 0, stream>>>(
            RX, KP_T, 0, UHP, 0, RH, KP_T, 0, N_NOTES, KP_T, NP_UH);
      }
      k_gru2<<<600, 256, 0, stream>>>(WZRH, BIAS, RH, ZB, X);
    }
    // ---- beat attention + tempo bilstms ----
    k_att_out<<<N_BEATS, 64, 0, stream>>>(X, fba_w, fba_b, fba_cv, res_info, BC);
    k_att_margin<<<N_BEATS, 64, 0, stream>>>(X, fma_w, fma_b, fma_cv, BC);
    k_bilstm<<<1, 256, 0, stream>>>(BC, 96, 83, N_BEATS, tr0f_wih, tr0f_whh, tr0f_b,
                                    tr0b_wih, tr0b_whh, tr0b_b, H1);
    k_bilstm<<<1, 256, 0, stream>>>(H1, 64, 64, N_BEATS, tr1f_wih, tr1f_whh, tr1f_b,
                                    tr1b_wih, tr1b_whh, tr1b_b, H2);
    k_tempo<<<CDIV(N_BEATS, 256), 256, 0, stream>>>(H2, tfc_w, tfc_b, TMPO);
    k_update_mid<<<500, 256, 0, stream>>>(beat_num, H2, X);
    // ---- final FC: hid = x @ fc_w1 (WMMA), other = relu(hid+b1) @ fc_w2 + b2 ----
    k_cvt_x<<<3750, 256, 0, stream>>>(X, XB);
    {
      int grps = mtiles * ((NP_FC1 / 16) / 4);  // 125
      k_wmma_gemmT<4><<<dim3(CDIV(grps, 4), 1, 1), 128, 0, stream>>>(
          XB, KP_X, 0, FC1P, 0, HID, NP_FC1, 0, N_NOTES, KP_X, NP_FC1);
    }
    k_other<<<N_NOTES, 64, 0, stream>>>(HID, fc_b1, fc_w2, fc_b2, TMPO, beat_num, X, out);
  }
}
